// Embedding_16432544875220
// MI455X (gfx1250) — matrix operand order repaired and checked by reading, not compiled, not run on hardware
//
#include <hip/hip_runtime.h>
#include <hip/hip_bf16.h>

// ---------------------------------------------------------------------------
// Shapes
// ---------------------------------------------------------------------------
#define BATCH   64
#define TVIS    200
#define TROWS   198            // MAX_VISIT - 2
#define NROWS   (BATCH * TROWS)   // 12672 (= 792 * 16)
#define PCODES  40
#define NCODE1  507            // NUM_CODE + 1
#define KTGT    100
#define DOTH    57
#define DFWD    664
#define LSTM    256

typedef __attribute__((ext_vector_type(16))) _Float16 v16h;
typedef __attribute__((ext_vector_type(8)))  float    v8f;

// ---------------------------------------------------------------------------
// K0a: per-batch tgt contribution  T[b,n] = sum_k tgt[b,k] * W[507+k, n]
//      tgt[b,k] = 1 - (weight_idx[k] in target[b,:])
// ---------------------------------------------------------------------------
__global__ void k_tgt(const int* __restrict__ target, const int* __restrict__ weight_idx,
                      const float* __restrict__ Wf, const float* __restrict__ Wb,
                      float* __restrict__ Tf, float* __restrict__ Tb)
{
    const int b = blockIdx.x;
    const int tid = threadIdx.x;
    __shared__ float tg[KTGT];
    if (tid < KTGT) {
        int wi = weight_idx[tid];
        float v = 1.0f;
        #pragma unroll
        for (int j = 0; j < 10; ++j)
            if (target[b * 10 + j] == wi) v = 0.0f;
        tg[tid] = v;
    }
    __syncthreads();
    const int n = tid;
    float af = 0.0f, ab = 0.0f;
    for (int k = 0; k < KTGT; ++k) {
        float v = tg[k];
        af = fmaf(v, Wf[(NCODE1 + k) * LSTM + n], af);
        ab = fmaf(v, Wb[(NCODE1 + k) * LSTM + n], ab);
    }
    Tf[b * LSTM + n] = af;
    Tb[b * LSTM + n] = ab;
}

// ---------------------------------------------------------------------------
// K0b: W1 (256x256 f32, row-major [k][n]) -> W1t f16 [n][k]
// ---------------------------------------------------------------------------
__global__ void k_w1t(const float* __restrict__ W1, _Float16* __restrict__ W1t)
{
    const int n = blockIdx.x;
    const int k = threadIdx.x;
    W1t[n * 256 + k] = (_Float16)W1[k * 256 + n];
}

// ---------------------------------------------------------------------------
// K1: x_forward / x_backward via sparse gather-accumulate.
//   Wf/Wb rows: [0..506]=code multi-hot, [507..606]=tgt (precomputed T),
//               [607..633]=others[:, :, 0:27], [634..653]=others[:, :, 37:57],
//               [654..663]=interval (fwd: others[:,t,27:37]; bwd: shifted reversed)
// ---------------------------------------------------------------------------
__global__ void k_xfb(const int* __restrict__ code, const float* __restrict__ others,
                      const int* __restrict__ length,
                      const float* __restrict__ Wf, const float* __restrict__ bf, const float* __restrict__ Tf,
                      const float* __restrict__ Wb, const float* __restrict__ bb, const float* __restrict__ Tb,
                      float* __restrict__ outF, float* __restrict__ outB)
{
    const int row = blockIdx.x;                 // 0..NROWS-1
    const int b = row / TROWS;
    const int t = row % TROWS;
    const bool bwd = (blockIdx.y != 0);
    const float* __restrict__ W    = bwd ? Wb : Wf;
    const float* __restrict__ bias = bwd ? bb : bf;
    const float* __restrict__ T    = bwd ? Tb : Tf;
    float* __restrict__ out        = bwd ? outB : outF;

    __shared__ int   codes[PCODES];
    __shared__ int   keep[PCODES];
    __shared__ float oth[47];
    __shared__ float itv[10];

    const int tid = threadIdx.x;
    const int L = length[b];
    int ts = t;
    if (bwd) ts = (t < L) ? (L - 1 - t) : t;    // reverse_sequence index

    if (tid < PCODES)
        codes[tid] = code[(b * TVIS + ts) * PCODES + tid];
    if (tid >= 64 && tid < 64 + 47) {           // dense feature cols (reordered)
        int j = tid - 64;
        int c = (j < 27) ? j : (j + 10);        // 0..26 -> 0..26 ; 27..46 -> 37..56
        oth[j] = others[(b * TVIS + ts) * DOTH + c];
    }
    if (tid >= 128 && tid < 138) {              // interval (10)
        int j = tid - 128;
        float v;
        if (!bwd) {
            v = others[(b * TVIS + t) * DOTH + 27 + j];
        } else if (t == 0) {
            v = 0.0f;
        } else {
            int t2 = t - 1;
            int ti = (t2 < L) ? (L - 1 - t2) : t2;
            v = others[(b * TVIS + ti) * DOTH + 27 + j];
        }
        itv[j] = v;
    }
    __syncthreads();
    if (tid < PCODES) {                         // dedup (multi_hot .set semantics)
        int k = 1, c = codes[tid];
        for (int q = 0; q < tid; ++q)
            if (codes[q] == c) { k = 0; break; }
        keep[tid] = k;
    }
    __syncthreads();

    const int n = tid;
    float acc = bias[n] + T[b * LSTM + n];
    #pragma unroll 8
    for (int p = 0; p < PCODES; ++p)
        if (keep[p]) acc += W[codes[p] * LSTM + n];
    #pragma unroll 4
    for (int j = 0; j < 47; ++j)
        acc = fmaf(oth[j], W[(607 + j) * LSTM + n], acc);
    #pragma unroll
    for (int j = 0; j < 10; ++j)
        acc = fmaf(itv[j], W[(654 + j) * LSTM + n], acc);
    out[row * LSTM + n] = acc;
}

// ---------------------------------------------------------------------------
// K2: layer-0 MLP via gather: H0 = relu(x @ W0 + b0)  (stored f16)
//   W0 rows: [0..506]=code(t+1), [507..526]=others[t+1,37:57],
//            [527..536]=others[t+1,27:37], [537..546]=others[t+2,27:37],
//            [547..554]=others[t+1,19:27]
// ---------------------------------------------------------------------------
__global__ void k_h0(const int* __restrict__ code, const float* __restrict__ others,
                     const float* __restrict__ W0, const float* __restrict__ b0,
                     _Float16* __restrict__ H0)
{
    const int row = blockIdx.x;
    const int b = row / TROWS;
    const int t = row % TROWS;
    __shared__ int   codes[PCODES];
    __shared__ int   keep[PCODES];
    __shared__ float d[48];
    const int tid = threadIdx.x;
    const int t1 = t + 1;

    if (tid < PCODES)
        codes[tid] = code[(b * TVIS + t1) * PCODES + tid];
    if (tid >= 64 && tid < 112) {
        int j = tid - 64;
        float v;
        if (j < 20)      v = others[(b * TVIS + t1) * DOTH + 37 + j];
        else if (j < 30) v = others[(b * TVIS + t1) * DOTH + 27 + (j - 20)];
        else if (j < 40) v = others[(b * TVIS + t + 2) * DOTH + 27 + (j - 30)];
        else             v = others[(b * TVIS + t1) * DOTH + 19 + (j - 40)];
        d[j] = v;
    }
    __syncthreads();
    if (tid < PCODES) {
        int k = 1, c = codes[tid];
        for (int q = 0; q < tid; ++q)
            if (codes[q] == c) { k = 0; break; }
        keep[tid] = k;
    }
    __syncthreads();

    const int n = tid;
    float acc = b0[n];
    #pragma unroll 8
    for (int p = 0; p < PCODES; ++p)
        if (keep[p]) acc += W0[codes[p] * LSTM + n];
    #pragma unroll 4
    for (int j = 0; j < 48; ++j)
        acc = fmaf(d[j], W0[(507 + j) * LSTM + n], acc);
    acc = fmaxf(acc, 0.0f);
    H0[row * LSTM + n] = (_Float16)acc;
}

// ---------------------------------------------------------------------------
// ds_swizzle XOR butterfly (group-of-32 mode: offset = xor<<10 | and_mask 0x1f)
// ---------------------------------------------------------------------------
template <int XOR>
__device__ __forceinline__ float swz_add(float s)
{
    int v = __builtin_amdgcn_ds_swizzle(__float_as_int(s), (XOR << 10) | 0x1f);
    return s + __int_as_float(v);
}

// ---------------------------------------------------------------------------
// K3: fv = normalize(H0 @ W1 + b1) * mask  via v_wmma_f32_16x16x32_f16.
// One block = 16 M-rows x 256 N-cols (8 waves, each wave 2 16x16 N-tiles).
// A as loaded here (16x32 f16): M=lane%16, lane/16 selects K half (0..15 / 16..31), VGPR v: K=2v,2v+1.
// B as loaded here (32x16 f16): lane%16=N, lane/16 selects K half (0..15 / 16..31),
//                       VGPR v packs K=2v (lo) and 2v+1 (hi) within the half.
// C layout: lane<16: VGPR r -> M=r ; lane>=16: M=8+r ; N=lane%16.
// ---------------------------------------------------------------------------
__global__ void k_gemm_fv(const _Float16* __restrict__ H0, const _Float16* __restrict__ W1t,
                          const float* __restrict__ b1, const int* __restrict__ length,
                          float* __restrict__ fv)
{
    const int m0   = blockIdx.x * 16;
    const int tid  = threadIdx.x;
    const int wave = tid >> 5;           // 0..7
    const int lane = tid & 31;
    const int l16  = lane & 15;
    const int hi   = lane >> 4;          // 0 or 1

    __shared__ float rsum[16];
    if (tid < 16) rsum[tid] = 0.0f;
    __syncthreads();

    v8f c0 = {}, c1 = {};
    const int mrow  = m0 + l16;              // A row held by this lane
    const int ncol0 = wave * 32 + l16;       // N for tile 0
    const int ncol1 = ncol0 + 16;            // N for tile 1

    #pragma unroll
    for (int k0 = 0; k0 < 256; k0 += 32) {
        v16h a;
        #pragma unroll
        for (int v = 0; v < 8; ++v) {
            int K = k0 + 2 * v + 16 * hi;
            a[2 * v]     = H0[mrow * 256 + K];
            a[2 * v + 1] = H0[mrow * 256 + K + 1];
        }
        v16h vb0, vb1;
        #pragma unroll
        for (int v = 0; v < 8; ++v) {
            int K = k0 + 2 * v + 16 * hi;
            vb0[2 * v]     = W1t[ncol0 * 256 + K];
            vb0[2 * v + 1] = W1t[ncol0 * 256 + K + 1];
            vb1[2 * v]     = W1t[ncol1 * 256 + K];
            vb1[2 * v + 1] = W1t[ncol1 * 256 + K + 1];
        }
        c0 = __builtin_amdgcn_wmma_f32_16x16x32_f16(false, a, false, vb0, (short)0, c0, false, false);
        c1 = __builtin_amdgcn_wmma_f32_16x16x32_f16(false, a, false, vb1, (short)0, c1, false, false);
    }

    // h = c + b1; per-row sum of squares: ds_swizzle butterfly over the 16
    // column-lanes, then one LDS atomic per (wave, row) to combine waves.
    float h0v[8], h1v[8];
    const float bv0 = b1[ncol0];
    const float bv1 = b1[ncol1];
    #pragma unroll
    for (int r = 0; r < 8; ++r) {
        h0v[r] = c0[r] + bv0;
        h1v[r] = c1[r] + bv1;
        float s = h0v[r] * h0v[r] + h1v[r] * h1v[r];
        s = swz_add<1>(s);
        s = swz_add<2>(s);
        s = swz_add<4>(s);
        s = swz_add<8>(s);
        if (l16 == 0) atomicAdd(&rsum[hi * 8 + r], s);
    }
    __syncthreads();

    #pragma unroll
    for (int r = 0; r < 8; ++r) {
        const int grow = m0 + hi * 8 + r;       // C-layout row
        const int bidx = grow / TROWS;
        const int tt   = grow % TROWS;
        const float mk = (tt < (length[bidx] - 2)) ? 1.0f : 0.0f;
        const float inv = __frsqrt_rn(rsum[hi * 8 + r]) * mk;
        fv[grow * 256 + ncol0] = h0v[r] * inv;
        fv[grow * 256 + ncol1] = h1v[r] * inv;
    }
}

// ---------------------------------------------------------------------------
// Launcher
// ---------------------------------------------------------------------------
extern "C" void kernel_launch(void* const* d_in, const int* in_sizes, int n_in,
                              void* d_out, int out_size, void* d_ws, size_t ws_size,
                              hipStream_t stream)
{
    const int*   code       = (const int*)  d_in[0];
    const float* others     = (const float*)d_in[1];
    const int*   length     = (const int*)  d_in[2];
    const int*   target     = (const int*)  d_in[3];
    const int*   weight_idx = (const int*)  d_in[4];
    const float* Wf         = (const float*)d_in[5];
    const float* bf         = (const float*)d_in[6];
    const float* Wb         = (const float*)d_in[7];
    const float* bb         = (const float*)d_in[8];
    const float* W0         = (const float*)d_in[9];
    const float* b0         = (const float*)d_in[10];
    const float* W1         = (const float*)d_in[11];
    const float* b1         = (const float*)d_in[12];

    float* outF = (float*)d_out;             // [12672, 256]
    float* outB = outF + (size_t)NROWS * LSTM;
    float* fv   = outB + (size_t)NROWS * LSTM;

    char* ws = (char*)d_ws;
    _Float16* H0  = (_Float16*)ws;                                   // 12672*256 f16
    _Float16* W1t = (_Float16*)(ws + (size_t)NROWS * LSTM * 2);      // 256*256 f16
    float*    Tf  = (float*)(ws + (size_t)NROWS * LSTM * 2 + 256 * 256 * 2);
    float*    Tb  = Tf + BATCH * LSTM;

    k_tgt<<<BATCH, 256, 0, stream>>>(target, weight_idx, Wf, Wb, Tf, Tb);
    k_w1t<<<256, 256, 0, stream>>>(W1, W1t);
    k_h0 <<<NROWS, 256, 0, stream>>>(code, others, W0, b0, H0);
    k_xfb<<<dim3(NROWS, 2), 256, 0, stream>>>(code, others, length,
                                              Wf, bf, Tf, Wb, bb, Tb, outF, outB);
    k_gemm_fv<<<NROWS / 16, 256, 0, stream>>>(H0, W1t, b1, length, fv);
}
